// BinomialDevianceLoss_89412629168723
// MI455X (gfx1250) — matrix operand order repaired and checked by reading, not compiled, not run on hardware
//
#include <hip/hip_runtime.h>
#include <hip/hip_bf16.h>

typedef __attribute__((ext_vector_type(16))) _Float16 v16h;
typedef __attribute__((ext_vector_type(8)))  _Float16 v8h;
typedef __attribute__((ext_vector_type(4)))  _Float16 v4h;
typedef __attribute__((ext_vector_type(8)))  float    v8f;

#define NROWS 8192
#define DIM   128

static constexpr float ALPHA = 2.0f;
static constexpr float BETA  = 0.5f;

// Wave32 xor-butterfly reduction via ds_swizzle (group-of-32 mode:
// offset = {1'b0, xor[14:10], or[9:5], and[4:0]}), SWAPX16..SWAPX1.
__device__ __forceinline__ float wave_reduce_sum(float v) {
    v += __int_as_float(__builtin_amdgcn_ds_swizzle(__float_as_int(v), 0x1F | (16 << 10)));
    v += __int_as_float(__builtin_amdgcn_ds_swizzle(__float_as_int(v), 0x1F | (8  << 10)));
    v += __int_as_float(__builtin_amdgcn_ds_swizzle(__float_as_int(v), 0x1F | (4  << 10)));
    v += __int_as_float(__builtin_amdgcn_ds_swizzle(__float_as_int(v), 0x1F | (2  << 10)));
    v += __int_as_float(__builtin_amdgcn_ds_swizzle(__float_as_int(v), 0x1F | (1  << 10)));
    return v;
}

// ---------------------------------------------------------------------------
// Kernel 1: row-normalize x and y into f16 (cosine => normalize then dot).
// One wave per row, 4 floats per lane. Rows [0,8192)->x, [8192,16384)->y.
// ---------------------------------------------------------------------------
__global__ void __launch_bounds__(256)
normalize_rows_f16(const float* __restrict__ x, const float* __restrict__ y,
                   _Float16* __restrict__ xn, _Float16* __restrict__ yn) {
    const int wave = (blockIdx.x * blockDim.x + threadIdx.x) >> 5;
    const int lane = threadIdx.x & 31;

    const float* src;
    _Float16*    dst;
    int row;
    if (wave < NROWS) { src = x; dst = xn; row = wave; }
    else              { src = y; dst = yn; row = wave - NROWS; }

    const float4 v = ((const float4*)(src + (size_t)row * DIM))[lane];
    float s = v.x * v.x + v.y * v.y + v.z * v.z + v.w * v.w;
    s = wave_reduce_sum(s);
    const float rinv = rsqrtf(s);

    v4h h;
    h[0] = (_Float16)(v.x * rinv);
    h[1] = (_Float16)(v.y * rinv);
    h[2] = (_Float16)(v.z * rinv);
    h[3] = (_Float16)(v.w * rinv);
    ((v4h*)(dst + (size_t)row * DIM))[lane] = h;
}

__global__ void zero_out(float* out) { out[0] = 0.0f; }

__device__ __forceinline__ float softplus_fast(float t) {
    // log1p(exp(t)), stable form, HW transcendentals (v_exp_f32 / v_log_f32)
    return fmaxf(t, 0.0f) + __logf(1.0f + __expf(-fabsf(t)));
}

// ---------------------------------------------------------------------------
// Kernel 2: one wave per 16x128 strip (8 consecutive 16x16 tiles) of the
// 8192x8192 score matrix. Tile indices are forced into SGPRs (readfirstlane)
// so tile-level branches are s_cbranch (EXEC untouched -> legal for WMMA).
//
// Fragment layouts (CDNA5 ISA 7.12.2, wave32):
//   A 16x32 f16 : lane holds row M=lane&15; halves 0..7 -> K=8*(lane>>4)+e,
//                 halves 8..15 -> K=16+8*(lane>>4)+e.
//   B 32x16 f16 : lane holds col N=lane&15; halves 0..7 -> K=8*(lane>>4)+e,
//                 halves 8..15 -> K=16+8*(lane>>4)+e.
//   C/D 16x16 f32: lane group gives N=lane&15, VGPR v -> M = v + 8*(lane>>4).
//
// m/w are streamed once (512 MB total) -> non-temporal loads so the 4 MB of
// normalized operands stay resident in the 192 MB L2.
// Tiles strictly below the block diagonal skip WMMA (S==0 there) but still
// stream m/w:  loss += w * softplus(-ALPHA*(0-BETA)*m) = w * softplus(m).
// ---------------------------------------------------------------------------
__global__ void __launch_bounds__(256)
cosine_triu_loss(const _Float16* __restrict__ xn, const _Float16* __restrict__ yn,
                 const float* __restrict__ m, const float* __restrict__ w,
                 float* __restrict__ out) {
    const int lane = threadIdx.x & 31;
    // Scalarize the wave id so it/jt/branches live in SGPRs.
    const int wid  = __builtin_amdgcn_readfirstlane(threadIdx.x >> 5);
    const int wgid = blockIdx.x * 8 + wid;
    const int it   = wgid >> 6;          // 0..511 : i-tile
    const int jt0  = (wgid & 63) << 3;   // first of 8 j-tiles
    const int r    = lane & 15;
    const int hi   = lane >> 4;
    const int ibase = it << 4;

    float acc = 0.0f;

    // A fragments: 4 chunks of K=32 (K total = 128), held for the whole strip.
    v16h a[4];
    if (jt0 + 7 >= it) {                 // scalar branch: strip touches triu
        const _Float16* arow = xn + (size_t)(ibase + r) * DIM;
        #pragma unroll
        for (int c = 0; c < 4; ++c) {
            const int k0 = 32 * c;
            const v8h lo = *(const v8h*)(arow + k0 + 8 * hi);
            const v8h hh = *(const v8h*)(arow + k0 + 16 + 8 * hi);
            #pragma unroll
            for (int e = 0; e < 8; ++e) { a[c][e] = lo[e]; a[c][8 + e] = hh[e]; }
        }
    }

    for (int t = 0; t < 8; ++t) {
        const int jt = jt0 + t;          // scalar
        const int j  = (jt << 4) + r;
        const bool gemm = (jt >= it);    // scalar condition

        // 1) B fragments first (feed WMMA as early as possible)
        v16h b[4];
        if (gemm) {
            const _Float16* brow = yn + (size_t)j * DIM;
            #pragma unroll
            for (int cc = 0; cc < 4; ++cc) {
                const v8h lo = *(const v8h*)(brow + 32 * cc + 8 * hi);
                const v8h hh = *(const v8h*)(brow + 32 * cc + 16 + 8 * hi);
                #pragma unroll
                for (int e = 0; e < 8; ++e) { b[cc][e] = lo[e]; b[cc][8 + e] = hh[e]; }
            }
        }

        // 2) m/w streaming loads (non-temporal), in flight during WMMA
        float mm[8], ww[8];
        #pragma unroll
        for (int v = 0; v < 8; ++v) {
            const int    i   = ibase + v + 8 * hi;
            const size_t idx = (size_t)i * NROWS + j;
            mm[v] = __builtin_nontemporal_load(m + idx);
            ww[v] = __builtin_nontemporal_load(w + idx);
        }

        // 3) WMMA chain (back-to-back, single wait on B)
        v8f c = {};
        if (gemm) {
            #pragma unroll
            for (int cc = 0; cc < 4; ++cc)
                c = __builtin_amdgcn_wmma_f32_16x16x32_f16(
                        false, a[cc], false, b[cc], (short)0, c, false, false);
        }

        // 4) masked elementwise loss
        #pragma unroll
        for (int v = 0; v < 8; ++v) {
            const int   i  = ibase + v + 8 * hi;
            const float s  = (j >= i) ? c[v] : 0.0f;
            const float tt = -ALPHA * (s - BETA) * mm[v];
            acc += ww[v] * softplus_fast(tt);
        }
    }

    acc = wave_reduce_sum(acc);
    if (lane == 0)
        atomicAdd(out, acc);
}

// ---------------------------------------------------------------------------
extern "C" void kernel_launch(void* const* d_in, const int* in_sizes, int n_in,
                              void* d_out, int out_size, void* d_ws, size_t ws_size,
                              hipStream_t stream) {
    const float* x = (const float*)d_in[0];   // [8192,128]
    const float* y = (const float*)d_in[1];   // [8192,128]
    const float* m = (const float*)d_in[2];   // [8192,8192]
    const float* w = (const float*)d_in[3];   // [8192,8192]
    float* out = (float*)d_out;

    _Float16* xn = (_Float16*)d_ws;                 // 2 MB
    _Float16* yn = xn + (size_t)NROWS * DIM;        // +2 MB (ws >= 4 MB)

    // 16384 rows, one wave each, 8 waves per 256-thread block -> 2048 blocks
    normalize_rows_f16<<<2048, 256, 0, stream>>>(x, y, xn, yn);

    zero_out<<<1, 1, 0, stream>>>(out);

    // 512 i-tiles * 64 strips = 32768 waves; 8 waves/block -> 4096 blocks
    cosine_triu_loss<<<4096, 256, 0, stream>>>(xn, yn, m, w, out);
}
